// EncoderSimilarityFixed_38388417692140
// MI455X (gfx1250) — compile-verified, matrix operand order checked by reading, not run on hardware
//
#include <hip/hip_runtime.h>
#include <hip/hip_bf16.h>
#include <math.h>

// ---------------------------------------------------------------------------
// CDNA5 (gfx1250) wave32 WMMA bf16 pipeline for SGRAF-style similarity.
// Every GEMM is A @ B^T with ISA table-order K fragments: 6x global_load_b128 +
// 2x v_wmma_f32_16x16x32_bf16 per 32-K step, no load guards.
// ---------------------------------------------------------------------------

typedef __attribute__((ext_vector_type(16))) __bf16 v16bf;
typedef __attribute__((ext_vector_type(8)))  float  v8f;
typedef __attribute__((ext_vector_type(4)))  unsigned int u32x4;

#define EPSF   1e-8f
#define NEGBIG -3.4e38f

__device__ __forceinline__ float wred_sum(float x) {
#pragma unroll
  for (int o = 16; o > 0; o >>= 1) x += __shfl_xor(x, o, 32);
  return x;
}
__device__ __forceinline__ float wred_max(float x) {
#pragma unroll
  for (int o = 16; o > 0; o >>= 1) x = fmaxf(x, __shfl_xor(x, o, 32));
  return x;
}

// ---------------------------------------------------------------------------
// Batched GEMM, C[b] = act((A[b] @ Bt[b]^T + bias) * scale)
//   A : M x K bf16 (lda, multiple of 8), rows tile-overread OK (slack+store guard)
//   Bt: N x K bf16 (ldb, multiple of 8), same
//   K : multiple of 32 (zero-pad data if logical K smaller)
// One wave -> 16x32 C tile (two WMMAs sharing the A fragment).
// ---------------------------------------------------------------------------
__global__ void k_gemm_nt(const __bf16* __restrict__ A,
                          const __bf16* __restrict__ Bt,
                          const float* __restrict__ bias,
                          float* __restrict__ C,
                          int M, int N, int K, int lda, int ldb, int ldc,
                          long long sA, long long sB, long long sC,
                          int act, float scale) {
  const int lane = threadIdx.x & 31;
  const int wave = threadIdx.x >> 5;
  const int mt = (M + 15) >> 4;
  const int nt2 = (N + 31) >> 5;
  const int tile = blockIdx.x * (blockDim.x >> 5) + wave;
  if (tile >= mt * nt2) return;
  const int tm = tile / nt2, tn = tile % nt2;
  const long long b = blockIdx.y;

  const int khalf = (lane >> 4) << 3;  // 0 or 8 (ISA 16-bit A/B lane layout)
  const __bf16* Ab  = A  + b * sA + (long long)((tm << 4) + (lane & 15)) * lda;
  const __bf16* Bb0 = Bt + b * sB + (long long)((tn << 5) + (lane & 15)) * ldb;
  const __bf16* Bb1 = Bb0 + (long long)16 * ldb;

  v8f acc0 = {0.f, 0.f, 0.f, 0.f, 0.f, 0.f, 0.f, 0.f};
  v8f acc1 = {0.f, 0.f, 0.f, 0.f, 0.f, 0.f, 0.f, 0.f};

  for (int k0 = 0; k0 < K; k0 += 32) {
    union { v16bf v; u32x4 q[2]; } af, bf0, bf1;
    const int o0 = k0 + khalf, o1 = k0 + khalf + 16;
    __builtin_prefetch(Ab + k0 + 256, 0, 1);
    af.q[0]  = *(const u32x4*)(Ab  + o0);
    af.q[1]  = *(const u32x4*)(Ab  + o1);
    bf0.q[0] = *(const u32x4*)(Bb0 + o0);
    bf0.q[1] = *(const u32x4*)(Bb0 + o1);
    bf1.q[0] = *(const u32x4*)(Bb1 + o0);
    bf1.q[1] = *(const u32x4*)(Bb1 + o1);
    acc0 = __builtin_amdgcn_wmma_f32_16x16x32_bf16(false, af.v, false, bf0.v,
                                                   (short)0, acc0, false, false);
    acc1 = __builtin_amdgcn_wmma_f32_16x16x32_bf16(false, af.v, false, bf1.v,
                                                   (short)0, acc1, false, false);
  }

  float* Cb = C + b * sC;
  const int crow0 = (tm << 4) + ((lane >> 4) << 3);
  const int col0 = (tn << 5) + (lane & 15);
  const int col1 = col0 + 16;
#pragma unroll
  for (int half = 0; half < 2; ++half) {
    const int col = half ? col1 : col0;
    if (col < N) {
      const float bvv = bias ? bias[col] : 0.f;
      const v8f acc = half ? acc1 : acc0;
#pragma unroll
      for (int j = 0; j < 8; ++j) {
        const int crow = crow0 + j;
        if (crow < M) {
          float y = (acc[j] + bvv) * scale;
          if (act == 1) y = tanhf(y);
          else if (act == 2) y = fmaxf(y, 0.f);
          Cb[(long long)crow * ldc + col] = y;
        }
      }
    }
  }
}

// ---------------------------------------------------------------------------
// f32 -> bf16 flat convert
__global__ void k_cvt(const float* __restrict__ s, __bf16* __restrict__ d, long long n) {
  long long i = (long long)blockIdx.x * blockDim.x + threadIdx.x;
  if (i < n) d[i] = (__bf16)s[i];
}

// f32 (K x N) -> bf16 transposed (N x K)
__global__ void k_cvt_T(const float* __restrict__ w, __bf16* __restrict__ wt, int K, int N) {
  long long i = (long long)blockIdx.x * blockDim.x + threadIdx.x;
  if (i >= (long long)K * N) return;
  int k = (int)(i / N), n = (int)(i % N);
  wt[(long long)n * K + k] = (__bf16)w[i];
}

// f32 -> bf16, 32-row batches, rows >= valid become exact zero
__global__ void k_cvt_pad32(const float* __restrict__ s, __bf16* __restrict__ d,
                            long long total, int rowlen, int valid) {
  long long i = (long long)blockIdx.x * blockDim.x + threadIdx.x;
  if (i >= total) return;
  int r = (int)((i / rowlen) % 32);
  d[i] = (r < valid) ? (__bf16)s[i] : (__bf16)0.f;
}

// imgT[n, d, r(64)] = r<36 ? img[n, r, d] : 0   (K-padded transposed image)
__global__ void k_imgT(const float* __restrict__ img, __bf16* __restrict__ imgT) {
  long long i = (long long)blockIdx.x * blockDim.x + threadIdx.x;
  if (i >= 128LL * 1024 * 64) return;
  int r = (int)(i & 63);
  long long t = i >> 6;
  int d = (int)(t & 1023);
  int n = (int)(t >> 10);
  imgT[i] = (r < 36) ? (__bf16)img[((long long)n * 36 + r) * 1024 + d] : (__bf16)0.f;
}

// simT[n, col, i(32)] = i<31 ? sim[n, i, col] : 0
__global__ void k_simT(const float* __restrict__ sim, __bf16* __restrict__ simT) {
  long long i = (long long)blockIdx.x * blockDim.x + threadIdx.x;
  if (i >= 128LL * 256 * 32) return;
  int r = (int)(i & 31);
  long long t = i >> 5;
  int col = (int)(t & 255);
  int n = (int)(t >> 8);
  simT[i] = (r < 31) ? (__bf16)sim[(long long)n * 32 * 256 + (long long)r * 256 + col]
                     : (__bf16)0.f;
}

// img_ave = mean over R=36
__global__ void k_rowmean(const float* __restrict__ img, float* __restrict__ ave) {
  int i = blockIdx.x * blockDim.x + threadIdx.x;
  if (i >= 128 * 1024) return;
  int n = i >> 10, dd = i & 1023;
  float a = 0.f;
  for (int r = 0; r < 36; ++r) a += img[((long long)(n * 36 + r) << 10) + dd];
  ave[i] = a * (1.f / 36.f);
}

// s[n,r] = dot(l_emb[n,r,:] * g_emb[n,:], v_cw) + v_cb   (wave per (n,r))
__global__ void k_img_s(const float* __restrict__ le, const float* __restrict__ ge,
                        const float* __restrict__ v_cw, const float* __restrict__ v_cb,
                        float* __restrict__ s_img) {
  int wid = blockIdx.x * (blockDim.x >> 5) + (threadIdx.x >> 5);
  int lane = threadIdx.x & 31;
  if (wid >= 128 * 36) return;
  int n = wid / 36;
  float acc = 0.f;
  for (int dd = lane; dd < 1024; dd += 32)
    acc += le[(long long)wid * 1024 + dd] * ge[(long long)n * 1024 + dd] * v_cw[dd];
  acc = wred_sum(acc);
  if (lane == 0) s_img[wid] = acc + v_cb[0];
}

// softmax over R, weighted sum, l2norm -> img_glo  (block per n)
__global__ void k_img_glo(const float* __restrict__ img, const float* __restrict__ s_img,
                          float* __restrict__ glo) {
  __shared__ float w[36];
  __shared__ float red[256];
  __shared__ float dsh;
  int n = blockIdx.x, tid = threadIdx.x;
  if (tid == 0) {
    float m = NEGBIG;
    for (int r = 0; r < 36; ++r) m = fmaxf(m, s_img[n * 36 + r]);
    float ssum = 0.f;
    for (int r = 0; r < 36; ++r) { float e = __expf(s_img[n * 36 + r] - m); w[r] = e; ssum += e; }
    for (int r = 0; r < 36; ++r) w[r] /= ssum;
  }
  __syncthreads();
  float ss = 0.f;
  for (int dd = tid; dd < 1024; dd += 256) {
    float g = 0.f;
    for (int r = 0; r < 36; ++r) g += w[r] * img[((long long)(n * 36 + r)) * 1024 + dd];
    glo[(long long)n * 1024 + dd] = g;
    ss += g * g;
  }
  red[tid] = ss; __syncthreads();
  for (int s2 = 128; s2 > 0; s2 >>= 1) { if (tid < s2) red[tid] += red[tid + s2]; __syncthreads(); }
  if (tid == 0) dsh = sqrtf(red[0]) + EPSF;
  __syncthreads();
  for (int dd = tid; dd < 1024; dd += 256) glo[(long long)n * 1024 + dd] /= dsh;
}

// masked caption mean (block per caption)
__global__ void k_cap_stats(const float* __restrict__ cap, const int* __restrict__ lens,
                            float* __restrict__ ave) {
  int c = blockIdx.x, tid = threadIdx.x;
  int len = lens[c];
  float inv = 1.f / (float)len;
  for (int dd = tid; dd < 1024; dd += 256) {
    float a = 0.f;
    for (int l = 0; l < len; ++l) a += cap[((long long)(c * 30 + l)) * 1024 + dd];
    ave[(long long)c * 1024 + dd] = a * inv;
  }
}

// caption attention weights + l2norm'd cap_glo  (block per caption)
__global__ void k_cap_tw_glo(const float* __restrict__ cap, const int* __restrict__ lens,
                             const float* __restrict__ tl, const float* __restrict__ tg,
                             const float* __restrict__ t_cw, const float* __restrict__ t_cb,
                             float* __restrict__ cap_glo) {
  __shared__ float red[256];
  __shared__ float tw[30];
  __shared__ float dsh;
  int c = blockIdx.x, tid = threadIdx.x;
  int len = lens[c];
  for (int l = 0; l < 30; ++l) {
    float p = 0.f;
    for (int dd = tid; dd < 1024; dd += 256)
      p += tl[((long long)(c * 30 + l)) * 1024 + dd] * tg[(long long)c * 1024 + dd] * t_cw[dd];
    red[tid] = p; __syncthreads();
    for (int s2 = 128; s2 > 0; s2 >>= 1) { if (tid < s2) red[tid] += red[tid + s2]; __syncthreads(); }
    if (tid == 0) tw[l] = red[0] + t_cb[0];
    __syncthreads();
  }
  if (tid == 0) {
    float m = NEGBIG;
    for (int l = 0; l < len; ++l) m = fmaxf(m, tw[l]);
    float s = 0.f;
    for (int l = 0; l < 30; ++l) { float e = (l < len) ? __expf(tw[l] - m) : 0.f; tw[l] = e; s += e; }
    for (int l = 0; l < 30; ++l) tw[l] /= s;
  }
  __syncthreads();
  float ss = 0.f;
  for (int dd = tid; dd < 1024; dd += 256) {
    float g = 0.f;
    for (int l = 0; l < 30; ++l) g += tw[l] * cap[((long long)(c * 30 + l)) * 1024 + dd];
    cap_glo[(long long)c * 1024 + dd] = g;
    ss += g * g;
  }
  red[tid] = ss; __syncthreads();
  for (int s2 = 128; s2 > 0; s2 >>= 1) { if (tid < s2) red[tid] += red[tid + s2]; __syncthreads(); }
  if (tid == 0) dsh = sqrtf(red[0]) + EPSF;
  __syncthreads();
  for (int dd = tid; dd < 1024; dd += 256) cap_glo[(long long)c * 1024 + dd] /= dsh;
}

// leaky-relu + mask + l2norm over L, store transposed f32.  (wave per (n,r))
__global__ void k_attn_post1(const float* __restrict__ attn, float* __restrict__ attnT,
                             const int* __restrict__ lens, int c) {
  int wid = blockIdx.x * (blockDim.x >> 5) + (threadIdx.x >> 5);
  int lane = threadIdx.x & 31;
  if (wid >= 128 * 36) return;
  int n = wid / 36, r = wid % 36;
  int len = lens[c];
  float v = 0.f;
  if (lane < 30 && lane < len) {
    float a = attn[(long long)wid * 30 + lane];
    v = (a > 0.f) ? a : 0.1f * a;
  }
  float ss = wred_sum(v * v);
  float denom = sqrtf(ss) + EPSF;
  if (lane < 30) attnT[((long long)n * 30 + lane) * 36 + r] = v / denom;
}

// softmax over R of SMOOTH*attnT, emit bf16 K-padded to 64 (wave per (n,l)).
// Output layout: [n, 32 rows (30 valid), 64 cols (36 valid, rest zero)].
__global__ void k_attn_post2(const float* __restrict__ attnT, __bf16* __restrict__ abf) {
  int wid = blockIdx.x * (blockDim.x >> 5) + (threadIdx.x >> 5);
  int lane = threadIdx.x & 31;
  if (wid >= 128 * 30) return;
  int n = wid / 30, l = wid % 30;
  long long sbase = (long long)wid * 36;
  long long dbase = ((long long)n * 32 + l) * 64;
  float x0 = 9.f * attnT[sbase + lane];
  int r1 = lane + 32;
  float x1 = (r1 < 36) ? 9.f * attnT[sbase + r1] : NEGBIG;
  float m = wred_max(fmaxf(x0, x1));
  float e0 = __expf(x0 - m);
  float e1 = (r1 < 36) ? __expf(x1 - m) : 0.f;
  float s = wred_sum(e0 + e1);
  abf[dbase + lane] = (__bf16)(e0 / s);
  abf[dbase + r1] = (r1 < 36) ? (__bf16)(e1 / s) : (__bf16)0.f;
}

// l2norm ctx row, then X = (ctx - cap)^2 as bf16.  (block per (n,l))
__global__ void k_ctx_post(const float* __restrict__ ctx, const float* __restrict__ cap,
                           int c, __bf16* __restrict__ X) {
  __shared__ float red[256];
  __shared__ float dsh;
  int row = blockIdx.x, tid = threadIdx.x;
  int l = row % 30;
  long long base = (long long)row * 1024;
  long long cb = (long long)(c * 30 + l) * 1024;
  float ss = 0.f;
  for (int dd = tid; dd < 1024; dd += 256) { float v = ctx[base + dd]; ss += v * v; }
  red[tid] = ss; __syncthreads();
  for (int s2 = 128; s2 > 0; s2 >>= 1) { if (tid < s2) red[tid] += red[tid + s2]; __syncthreads(); }
  if (tid == 0) dsh = sqrtf(red[0]) + EPSF;
  __syncthreads();
  for (int dd = tid; dd < 1024; dd += 256) {
    float v = ctx[base + dd] / dsh - cap[cb + dd];
    X[base + dd] = (__bf16)(v * v);
  }
}

// sim_glo row (block per n, thread per output col)
__global__ void k_sim_glo(const float* __restrict__ ig, const float* __restrict__ cg, int c,
                          const float* __restrict__ gw, const float* __restrict__ gb,
                          float* __restrict__ sim) {
  __shared__ float red[256];
  __shared__ float dsh;
  int n = blockIdx.x, t = threadIdx.x;
  float acc = 0.f;
  for (int dd = 0; dd < 1024; ++dd) {
    float df = ig[(long long)n * 1024 + dd] - cg[(long long)c * 1024 + dd];
    acc += df * df * gw[(long long)dd * 256 + t];
  }
  acc += gb[t];
  red[t] = acc * acc; __syncthreads();
  for (int s2 = 128; s2 > 0; s2 >>= 1) { if (t < s2) red[t] += red[t + s2]; __syncthreads(); }
  if (t == 0) dsh = sqrtf(red[0]) + EPSF;
  __syncthreads();
  sim[(long long)n * 32 * 256 + t] = acc / dsh;
}

// l2norm sim_loc rows (block per (n,l)); rows 1..30 of sim_emb
__global__ void k_row_l2norm(float* __restrict__ sim) {
  __shared__ float red[256];
  __shared__ float dsh;
  int idx = blockIdx.x;
  int n = idx / 30, l = idx % 30;
  float* row = sim + (long long)n * 32 * 256 + (long long)(1 + l) * 256;
  int t = threadIdx.x;
  float v = row[t];
  red[t] = v * v; __syncthreads();
  for (int s2 = 128; s2 > 0; s2 >>= 1) { if (t < s2) red[t] += red[t + s2]; __syncthreads(); }
  if (t == 0) dsh = sqrtf(red[0]) + EPSF;
  __syncthreads();
  row[t] = v / dsh;
}

// masked softmax of SGR attention rows -> bf16 padded 32x32 (wave per (n,i))
__global__ void k_e_softmax(const float* __restrict__ e, __bf16* __restrict__ ebf,
                            const int* __restrict__ lens, int c) {
  int wid = blockIdx.x * (blockDim.x >> 5) + (threadIdx.x >> 5);
  int lane = threadIdx.x & 31;
  if (wid >= 128 * 32) return;
  int i = wid & 31;
  int len = lens[c];
  bool valid = (lane < 31) && ((lane == 0) || (lane - 1) < len);
  float x = valid ? e[(long long)wid * 32 + lane] : NEGBIG;
  float m = wred_max(x);
  float ex = valid ? __expf(x - m) : 0.f;
  float s = wred_sum(ex);
  float r = (i < 31) ? ex / s : 0.f;
  ebf[(long long)wid * 32 + lane] = (__bf16)r;
}

// eval head: out[n, c] = dot(sim[n,0,:], eval_w) + eval_b   (wave per n)
__global__ void k_eval(const float* __restrict__ sim, const float* __restrict__ ew,
                       const float* __restrict__ eb, float* __restrict__ out, int c) {
  int wid = blockIdx.x * (blockDim.x >> 5) + (threadIdx.x >> 5);
  int lane = threadIdx.x & 31;
  if (wid >= 128) return;
  float acc = 0.f;
  for (int t = lane; t < 256; t += 32) acc += sim[(long long)wid * 32 * 256 + t] * ew[t];
  acc = wred_sum(acc);
  if (lane == 0) out[wid * 32 + c] = acc + eb[0];
}

// ---------------------------------------------------------------------------
extern "C" void kernel_launch(void* const* d_in, const int* in_sizes, int n_in,
                              void* d_out, int out_size, void* d_ws, size_t ws_size,
                              hipStream_t stream) {
  (void)in_sizes; (void)n_in; (void)out_size; (void)ws_size;
  const float* img_emb = (const float*)d_in[0];
  const float* cap_emb = (const float*)d_in[1];
  const int*   cap_lens = (const int*)d_in[2];
  const float* v_lw = (const float*)d_in[3];  const float* v_lb = (const float*)d_in[4];
  const float* v_gw = (const float*)d_in[5];  const float* v_gb = (const float*)d_in[6];
  const float* v_cw = (const float*)d_in[7];  const float* v_cb = (const float*)d_in[8];
  const float* t_lw = (const float*)d_in[9];  const float* t_lb = (const float*)d_in[10];
  const float* t_gw = (const float*)d_in[11]; const float* t_gb = (const float*)d_in[12];
  const float* t_cw = (const float*)d_in[13]; const float* t_cb = (const float*)d_in[14];
  const float* loc_w = (const float*)d_in[15]; const float* loc_b = (const float*)d_in[16];
  const float* glo_w = (const float*)d_in[17]; const float* glo_b = (const float*)d_in[18];
  const float* eval_w = (const float*)d_in[19]; const float* eval_b = (const float*)d_in[20];
  const float* sgr_qw = (const float*)d_in[21]; const float* sgr_qb = (const float*)d_in[22];
  const float* sgr_kw = (const float*)d_in[23]; const float* sgr_kb = (const float*)d_in[24];
  const float* sgr_gw = (const float*)d_in[25]; const float* sgr_gb = (const float*)d_in[26];
  float* out = (float*)d_out;

  const float BN_SCALE = 1.0f / sqrtf(1.0f + 1e-5f);

  char* base = (char*)d_ws;
  size_t off = 0;
  auto alloc = [&](size_t bytes) -> void* {
    off = (off + 255) & ~(size_t)255;
    void* p = base + off;
    off += bytes;
    return p;
  };

  __bf16* img_bf   = (__bf16*)alloc(128LL * 36 * 1024 * 2);
  __bf16* cap_bf   = (__bf16*)alloc(32LL * 30 * 1024 * 2);
  __bf16* vlwT     = (__bf16*)alloc(1024LL * 1024 * 2);
  __bf16* vgwT     = (__bf16*)alloc(1024LL * 1024 * 2);
  __bf16* tlwT     = (__bf16*)alloc(1024LL * 1024 * 2);
  __bf16* tgwT     = (__bf16*)alloc(1024LL * 1024 * 2);
  __bf16* locwT    = (__bf16*)alloc(256LL * 1024 * 2);
  __bf16* qwT      = (__bf16*)alloc(3LL * 256 * 256 * 2);
  __bf16* kwT      = (__bf16*)alloc(3LL * 256 * 256 * 2);
  __bf16* gwT      = (__bf16*)alloc(3LL * 256 * 256 * 2);
  __bf16* imgT     = (__bf16*)alloc(128LL * 1024 * 64 * 2);
  float*  img_ave  = (float*)alloc(128LL * 1024 * 4);
  __bf16* iave_bf  = (__bf16*)alloc(128LL * 1024 * 2);
  float*  g_emb    = (float*)alloc(128LL * 1024 * 4);
  float*  l_emb    = (float*)alloc(4608LL * 1024 * 4);
  float*  s_img    = (float*)alloc(128LL * 36 * 4);
  float*  img_glo  = (float*)alloc(128LL * 1024 * 4);
  float*  cap_ave  = (float*)alloc(32LL * 1024 * 4);
  __bf16* cave_bf  = (__bf16*)alloc(32LL * 1024 * 2);
  float*  tlb      = (float*)alloc(960LL * 1024 * 4);
  float*  tgb      = (float*)alloc(32LL * 1024 * 4);
  float*  cap_glo  = (float*)alloc(32LL * 1024 * 4);
  float*  attn     = (float*)alloc(128LL * 36 * 30 * 4);
  float*  attnT    = (float*)alloc(128LL * 30 * 36 * 4);
  __bf16* attnT_bf = (__bf16*)alloc(128LL * 32 * 64 * 2);
  float*  ctx      = (float*)alloc(128LL * 30 * 1024 * 4);
  __bf16* X_bf     = (__bf16*)alloc(128LL * 30 * 1024 * 2);
  float*  sim_emb  = (float*)alloc(128LL * 32 * 256 * 4);
  __bf16* sim_bf   = (__bf16*)alloc(128LL * 32 * 256 * 2);
  __bf16* simT     = (__bf16*)alloc(128LL * 256 * 32 * 2);
  float*  qf       = (float*)alloc(128LL * 32 * 256 * 4);
  float*  kf       = (float*)alloc(128LL * 32 * 256 * 4);
  float*  of       = (float*)alloc(128LL * 32 * 256 * 4);
  __bf16* q_bf     = (__bf16*)alloc(128LL * 32 * 256 * 2);
  __bf16* k_bf     = (__bf16*)alloc(128LL * 32 * 256 * 2);
  __bf16* o_bf     = (__bf16*)alloc(128LL * 32 * 256 * 2);
  float*  ebuf     = (float*)alloc(128LL * 32 * 32 * 4);
  __bf16* e_bf     = (__bf16*)alloc(128LL * 32 * 32 * 2);
  (void)alloc(262144);  // tail slack: tile overreach stays inside workspace

  auto cvt = [&](const float* s, __bf16* d, long long n) {
    k_cvt<<<dim3((unsigned)((n + 255) / 256)), dim3(256), 0, stream>>>(s, d, n);
  };
  auto cvtT = [&](const float* s, __bf16* d, int K, int N) {
    long long n = (long long)K * N;
    k_cvt_T<<<dim3((unsigned)((n + 255) / 256)), dim3(256), 0, stream>>>(s, d, K, N);
  };
  auto cvtpad = [&](const float* s, __bf16* d) {
    long long total = 128LL * 32 * 256;
    k_cvt_pad32<<<dim3((unsigned)((total + 255) / 256)), dim3(256), 0, stream>>>(s, d, total, 256, 31);
  };
  auto gemm = [&](const __bf16* A, const __bf16* Bt, const float* bias, float* C,
                  int M, int N, int K, int lda, int ldb, int ldc,
                  long long sA, long long sB, long long sC, int batch,
                  int act, float scale) {
    int tiles = ((M + 15) / 16) * ((N + 31) / 32);
    dim3 grid((unsigned)((tiles + 7) / 8), (unsigned)batch);
    k_gemm_nt<<<grid, dim3(256), 0, stream>>>(A, Bt, bias, C, M, N, K, lda, ldb, ldc,
                                              sA, sB, sC, act, scale);
  };

  // ---- prologue: bf16 conversions / transposes ----
  cvt(img_emb, img_bf, 128LL * 36 * 1024);
  cvt(cap_emb, cap_bf, 32LL * 30 * 1024);
  cvtT(v_lw, vlwT, 1024, 1024);
  cvtT(v_gw, vgwT, 1024, 1024);
  cvtT(t_lw, tlwT, 1024, 1024);
  cvtT(t_gw, tgwT, 1024, 1024);
  cvtT(loc_w, locwT, 1024, 256);
  for (int s = 0; s < 3; ++s) {
    cvtT(sgr_qw + (long long)s * 65536, qwT + (long long)s * 65536, 256, 256);
    cvtT(sgr_kw + (long long)s * 65536, kwT + (long long)s * 65536, 256, 256);
    cvtT(sgr_gw + (long long)s * 65536, gwT + (long long)s * 65536, 256, 256);
  }
  k_imgT<<<dim3((unsigned)((128LL * 1024 * 64 + 255) / 256)), dim3(256), 0, stream>>>(img_emb, imgT);

  // ---- image path ----
  k_rowmean<<<dim3(512), dim3(256), 0, stream>>>(img_emb, img_ave);
  cvt(img_ave, iave_bf, 128LL * 1024);
  gemm(iave_bf, vgwT, v_gb, g_emb, 128, 1024, 1024, 1024, 1024, 1024, 0, 0, 0, 1, 1, BN_SCALE);
  gemm(img_bf, vlwT, v_lb, l_emb, 4608, 1024, 1024, 1024, 1024, 1024, 0, 0, 0, 1, 1, BN_SCALE);
  k_img_s<<<dim3(128 * 36 / 8), dim3(256), 0, stream>>>(l_emb, g_emb, v_cw, v_cb, s_img);
  k_img_glo<<<dim3(128), dim3(256), 0, stream>>>(img_emb, s_img, img_glo);

  // ---- caption path ----
  k_cap_stats<<<dim3(32), dim3(256), 0, stream>>>(cap_emb, cap_lens, cap_ave);
  cvt(cap_ave, cave_bf, 32LL * 1024);
  gemm(cap_bf, tlwT, t_lb, tlb, 960, 1024, 1024, 1024, 1024, 1024, 0, 0, 0, 1, 1, 1.f);
  gemm(cave_bf, tgwT, t_gb, tgb, 32, 1024, 1024, 1024, 1024, 1024, 0, 0, 0, 1, 1, 1.f);
  k_cap_tw_glo<<<dim3(32), dim3(256), 0, stream>>>(cap_emb, cap_lens, tlb, tgb, t_cw, t_cb, cap_glo);

  // ---- per caption ----
  for (int c = 0; c < 32; ++c) {
    // attn[n] = img[n] (36x1024) @ cap_c^T
    gemm(img_bf, cap_bf + (long long)c * 30 * 1024, nullptr, attn,
         36, 30, 1024, 1024, 1024, 30, 36LL * 1024, 0, 36LL * 30, 128, 0, 1.f);
    k_attn_post1<<<dim3(128 * 36 / 8), dim3(256), 0, stream>>>(attn, attnT, cap_lens, c);
    k_attn_post2<<<dim3(128 * 30 / 8), dim3(256), 0, stream>>>(attnT, attnT_bf);
    // ctx[n] = attnT (30x64, K-padded) @ imgT[n]^T (1024x64)
    gemm(attnT_bf, imgT, nullptr, ctx,
         30, 1024, 64, 64, 64, 1024, 32LL * 64, 1024LL * 64, 30LL * 1024, 128, 0, 1.f);
    k_ctx_post<<<dim3(128 * 30), dim3(256), 0, stream>>>(ctx, cap_emb, c, X_bf);
    // sim_loc -> rows 1..30 of sim_emb
    gemm(X_bf, locwT, loc_b, sim_emb + 256,
         30, 256, 1024, 1024, 1024, 256, 30LL * 1024, 0, 32LL * 256, 128, 0, 1.f);
    k_sim_glo<<<dim3(128), dim3(256), 0, stream>>>(img_glo, cap_glo, c, glo_w, glo_b, sim_emb);
    k_row_l2norm<<<dim3(128 * 30), dim3(256), 0, stream>>>(sim_emb);

    for (int step = 0; step < 3; ++step) {
      cvtpad(sim_emb, sim_bf);
      k_simT<<<dim3((unsigned)((128LL * 256 * 32 + 255) / 256)), dim3(256), 0, stream>>>(sim_emb, simT);
      gemm(sim_bf, qwT + (long long)step * 65536, sgr_qb + step * 256, qf,
           31, 256, 256, 256, 256, 256, 32LL * 256, 0, 32LL * 256, 128, 0, 1.f);
      gemm(sim_bf, kwT + (long long)step * 65536, sgr_kb + step * 256, kf,
           31, 256, 256, 256, 256, 256, 32LL * 256, 0, 32LL * 256, 128, 0, 1.f);
      cvtpad(qf, q_bf);
      cvtpad(kf, k_bf);
      // e = q @ k^T (31x31)
      gemm(q_bf, k_bf, nullptr, ebuf,
           31, 31, 256, 256, 256, 32, 32LL * 256, 32LL * 256, 32LL * 32, 128, 0, 1.f);
      k_e_softmax<<<dim3(128 * 32 / 8), dim3(256), 0, stream>>>(ebuf, e_bf, cap_lens, c);
      // o = softmax(e) (32x32, zero-padded) @ simT^T
      gemm(e_bf, simT, nullptr, of,
           31, 256, 32, 32, 32, 256, 32LL * 32, 256LL * 32, 32LL * 256, 128, 0, 1.f);
      cvtpad(of, o_bf);
      // sim = relu(o @ gw^T + gb)
      gemm(o_bf, gwT + (long long)step * 65536, sgr_gb + step * 256, sim_emb,
           31, 256, 256, 256, 256, 256, 32LL * 256, 0, 32LL * 256, 128, 2, 1.f);
    }
    k_eval<<<dim3(16), dim3(256), 0, stream>>>(sim_emb, eval_w, eval_b, out, c);
  }
}
